// UniVForecaster_89936615178830
// MI455X (gfx1250) — compile-verified, matrix operand order checked by reading, not run on hardware
//
#include <hip/hip_runtime.h>

// ---------------------------------------------------------------------------
// Mamba univariate forecaster for MI455X (gfx1250, wave32, WMMA).
// Big GEMMs use v_wmma_f32_16x16x32_f16 (f16 in, f32 accumulate).
// Each GEMM wave computes a 32x64 C tile (2 A-frags x 4 B-frags = 8 WMMAs
// per K=32 step) for 21.3 FLOP/byte-of-load arithmetic intensity.
// ---------------------------------------------------------------------------

typedef __attribute__((ext_vector_type(16))) _Float16 v16h;
typedef __attribute__((ext_vector_type(8)))  _Float16 v8h;
typedef __attribute__((ext_vector_type(8)))  float    v8f;

#define B_      8
#define L_      512
#define DM      512      // D_MODEL
#define DI      1024     // D_INNER
#define DS      16       // D_STATE
#define DTR     32       // DT_RANK
#define NLAYER  4
#define HZN     96
#define MTOT    (B_ * L_)   // 4096 tokens

__device__ __forceinline__ float silu_f(float x) { return x / (1.0f + __expf(-x)); }

// ------------------------------- elementwise -------------------------------

__global__ void k_f2h(const float* __restrict__ s, _Float16* __restrict__ d, int n) {
    int i = blockIdx.x * blockDim.x + threadIdx.x;
    if (i < n) d[i] = (_Float16)s[i];
}

__global__ void k_embed(const float* __restrict__ x, const float* __restrict__ ew,
                        const float* __restrict__ eb, float* __restrict__ h) {
    int i = blockIdx.x * blockDim.x + threadIdx.x;   // over MTOT*DM
    int m = i >> 9;          // token (b*512 + t); x is (B,1,L) flat == m
    int d = i & (DM - 1);
    h[i] = x[m] * ew[d] + eb[d];
}

// RMSNorm: one block (256 thr) per row of 512, output f16 (GEMM A-matrix).
__global__ void k_rmsnorm(const float* __restrict__ h, const float* __restrict__ w,
                          _Float16* __restrict__ out) {
    __shared__ float red[256];
    int row = blockIdx.x, tid = threadIdx.x;
    const float* hr = h + (size_t)row * DM;
    float v0 = hr[tid], v1 = hr[tid + 256];
    red[tid] = v0 * v0 + v1 * v1;
    __syncthreads();
    for (int s = 128; s > 0; s >>= 1) {
        if (tid < s) red[tid] += red[tid + s];
        __syncthreads();
    }
    float rstd = rsqrtf(red[0] * (1.0f / DM) + 1e-5f);
    out[(size_t)row * DM + tid]       = (_Float16)(v0 * rstd * w[tid]);
    out[(size_t)row * DM + tid + 256] = (_Float16)(v1 * rstd * w[tid + 256]);
}

// --------------------------------- WMMA GEMM --------------------------------
// C[m,n] = sum_k A[m,k] * W[n,k]  (+ res[m,n])
// A: (M,K) f16 row-major.  W: (N,K) f16 row-major (weights as stored).
// One wave computes a 32x64 tile of C; K stepped by 32.
//
// Per ISA 16-bit fragment layout (wave32): lane l holds row/col (l&15);
// with koff = (l>>4)*8, halves 0..7 hold K = koff+0..7 and halves 8..15
// hold K = koff+16..23  -> two contiguous 16-byte loads per fragment.

__device__ __forceinline__ v16h ld_frag(const _Float16* p) {
    v8h lo = *(const v8h*)(p);
    v8h hi = *(const v8h*)(p + 16);
    v16h f;
#pragma unroll
    for (int i = 0; i < 8; ++i) { f[i] = lo[i]; f[8 + i] = hi[i]; }
    return f;
}

__global__ void k_gemm_nt(const _Float16* __restrict__ A, const _Float16* __restrict__ W,
                          const float* __restrict__ res, float* __restrict__ C,
                          int N, int K) {
    const int l     = threadIdx.x;           // 0..31, one wave per block
    const int mBase = blockIdx.x * 32;
    const int nBase = blockIdx.y * 64;
    const int r     = l & 15;
    const int koff  = (l >> 4) << 3;

    const _Float16* a0 = A + (size_t)(mBase +  0 + r) * K + koff;
    const _Float16* a1 = A + (size_t)(mBase + 16 + r) * K + koff;
    const _Float16* w0 = W + (size_t)(nBase +  0 + r) * K + koff;
    const _Float16* w1 = W + (size_t)(nBase + 16 + r) * K + koff;
    const _Float16* w2 = W + (size_t)(nBase + 32 + r) * K + koff;
    const _Float16* w3 = W + (size_t)(nBase + 48 + r) * K + koff;

    v8f c00 = {}, c01 = {}, c02 = {}, c03 = {};
    v8f c10 = {}, c11 = {}, c12 = {}, c13 = {};
    for (int k0 = 0; k0 < K; k0 += 32) {
        v16h fa0 = ld_frag(a0 + k0);
        v16h fa1 = ld_frag(a1 + k0);
        v16h b0  = ld_frag(w0 + k0);
        v16h b1  = ld_frag(w1 + k0);
        v16h b2  = ld_frag(w2 + k0);
        v16h b3  = ld_frag(w3 + k0);
        c00 = __builtin_amdgcn_wmma_f32_16x16x32_f16(false, fa0, false, b0, (short)0, c00, false, false);
        c01 = __builtin_amdgcn_wmma_f32_16x16x32_f16(false, fa0, false, b1, (short)0, c01, false, false);
        c02 = __builtin_amdgcn_wmma_f32_16x16x32_f16(false, fa0, false, b2, (short)0, c02, false, false);
        c03 = __builtin_amdgcn_wmma_f32_16x16x32_f16(false, fa0, false, b3, (short)0, c03, false, false);
        c10 = __builtin_amdgcn_wmma_f32_16x16x32_f16(false, fa1, false, b0, (short)0, c10, false, false);
        c11 = __builtin_amdgcn_wmma_f32_16x16x32_f16(false, fa1, false, b1, (short)0, c11, false, false);
        c12 = __builtin_amdgcn_wmma_f32_16x16x32_f16(false, fa1, false, b2, (short)0, c12, false, false);
        c13 = __builtin_amdgcn_wmma_f32_16x16x32_f16(false, fa1, false, b3, (short)0, c13, false, false);
    }

    // C/D layout: lane l -> n = nBase + (l&15) (+16*sub); vgpr i -> m = base + (l>>4)*8 + i
    const int mRow = mBase + ((l >> 4) << 3);
    const int n    = nBase + r;
#pragma unroll
    for (int i = 0; i < 8; ++i) {
        size_t o0 = (size_t)(mRow + i) * N + n;        // rows mBase..mBase+15
        size_t o1 = (size_t)(mRow + 16 + i) * N + n;   // rows mBase+16..mBase+31
        if (res) {
            C[o0]      = res[o0]      + c00[i];
            C[o0 + 16] = res[o0 + 16] + c01[i];
            C[o0 + 32] = res[o0 + 32] + c02[i];
            C[o0 + 48] = res[o0 + 48] + c03[i];
            C[o1]      = res[o1]      + c10[i];
            C[o1 + 16] = res[o1 + 16] + c11[i];
            C[o1 + 32] = res[o1 + 32] + c12[i];
            C[o1 + 48] = res[o1 + 48] + c13[i];
        } else {
            C[o0] = c00[i]; C[o0 + 16] = c01[i]; C[o0 + 32] = c02[i]; C[o0 + 48] = c03[i];
            C[o1] = c10[i]; C[o1 + 16] = c11[i]; C[o1 + 32] = c12[i]; C[o1 + 48] = c13[i];
        }
    }
}

// ----------------------------- conv / dt / scan -----------------------------

// Causal depthwise conv (D_CONV=4) over the x half of xz, then SiLU.
__global__ void k_conv_silu(const float* __restrict__ xz, const float* __restrict__ cw,
                            const float* __restrict__ cb, float* __restrict__ xin,
                            _Float16* __restrict__ xin16) {
    int i = blockIdx.x * blockDim.x + threadIdx.x;   // over MTOT*DI
    int m = i >> 10;
    int d = i & (DI - 1);
    int t = m & (L_ - 1);
    float acc = cb[d];
    const float* cwd = cw + d * 4;
#pragma unroll
    for (int j = 0; j < 4; ++j) {
        int tt = t - 3 + j;
        if (tt >= 0) acc += cwd[j] * xz[(size_t)(m - 3 + j) * (2 * DI) + d];
    }
    float v = silu_f(acc);
    xin[i] = v;
    xin16[i] = (_Float16)v;
}

// delta = softplus(dbc[:, :32] @ dt_w^T + dt_b)   (K=32, plain FMA)
__global__ void k_dt(const float* __restrict__ dbc, const float* __restrict__ dtw,
                     const float* __restrict__ dtb, float* __restrict__ delta) {
    int i = blockIdx.x * blockDim.x + threadIdx.x;   // over MTOT*DI
    int m = i >> 10;
    int d = i & (DI - 1);
    float acc = dtb[d];
    const float* wrow = dtw + (size_t)d * DTR;
    const float* drow = dbc + (size_t)m * 64;
#pragma unroll
    for (int r2 = 0; r2 < DTR; ++r2) acc += wrow[r2] * drow[r2];
    delta[i] = (acc > 20.0f) ? acc : log1pf(__expf(acc));
}

// Selective scan: grid (B, DI/256), 16 states per thread in registers,
// B/C rows broadcast through LDS each timestep.
__global__ void k_scan(const float* __restrict__ dbc, const float* __restrict__ delta,
                       const float* __restrict__ xin, const float* __restrict__ Alog,
                       const float* __restrict__ Dp, float* __restrict__ y) {
    const int b = blockIdx.x;
    const int d = blockIdx.y * 256 + threadIdx.x;
    float Ac[DS], st[DS];
#pragma unroll
    for (int s = 0; s < DS; ++s) { Ac[s] = -__expf(Alog[(size_t)d * DS + s]); st[s] = 0.0f; }
    const float Dd = Dp[d];
    __shared__ float sB[DS], sC[DS];
    for (int t = 0; t < L_; ++t) {
        const int m = (b << 9) + t;
        if (threadIdx.x < 2 * DS) {
            float v = dbc[(size_t)m * 64 + DTR + threadIdx.x];
            if (threadIdx.x < DS) sB[threadIdx.x] = v;
            else                  sC[threadIdx.x - DS] = v;
        }
        __syncthreads();
        const float dl = delta[(size_t)m * DI + d];
        const float xv = xin[(size_t)m * DI + d];
        float acc = 0.0f;
#pragma unroll
        for (int s = 0; s < DS; ++s) {
            st[s] = __expf(dl * Ac[s]) * st[s] + dl * sB[s] * xv;
            acc += st[s] * sC[s];
        }
        y[(size_t)m * DI + d] = acc + Dd * xv;
        __syncthreads();
    }
}

// y = y * silu(z), emitted as f16 (A-matrix of out_proj GEMM)
__global__ void k_gate(const float* __restrict__ xz, const float* __restrict__ y,
                       _Float16* __restrict__ y16) {
    int i = blockIdx.x * blockDim.x + threadIdx.x;   // over MTOT*DI
    int m = i >> 10;
    int d = i & (DI - 1);
    float z = xz[(size_t)m * (2 * DI) + DI + d];
    y16[i] = (_Float16)(y[i] * silu_f(z));
}

__global__ void k_head(const float* __restrict__ h, const float* __restrict__ hw,
                       const float* __restrict__ hb, float* __restrict__ out) {
    int i = blockIdx.x * blockDim.x + threadIdx.x;
    if (i >= B_ * HZN) return;
    int b = i / HZN, j = i % HZN;
    const float* hr = h + ((size_t)(b * L_ + (L_ - 1))) * DM;
    const float* wr = hw + (size_t)j * DM;
    float acc = hb[j];
    for (int d = 0; d < DM; ++d) acc += hr[d] * wr[d];
    out[i] = acc;
}

// --------------------------------- launcher ---------------------------------

extern "C" void kernel_launch(void* const* d_in, const int* in_sizes, int n_in,
                              void* d_out, int out_size, void* d_ws, size_t ws_size,
                              hipStream_t stream) {
    (void)in_sizes; (void)n_in; (void)out_size; (void)ws_size;

    const float* x         = (const float*)d_in[0];
    const float* embed_w   = (const float*)d_in[1];
    const float* embed_b   = (const float*)d_in[2];
    const float* norm_w    = (const float*)d_in[3];
    const float* in_proj_w = (const float*)d_in[4];
    const float* conv_w    = (const float*)d_in[5];
    const float* conv_b    = (const float*)d_in[6];
    const float* x_proj_w  = (const float*)d_in[7];
    const float* dt_w      = (const float*)d_in[8];
    const float* dt_b      = (const float*)d_in[9];
    const float* A_log     = (const float*)d_in[10];
    const float* Dvec      = (const float*)d_in[11];
    const float* out_proj_w= (const float*)d_in[12];
    const float* head_w    = (const float*)d_in[13];
    const float* head_b    = (const float*)d_in[14];
    float* out = (float*)d_out;

    char* p = (char*)d_ws;
    float*    h      = (float*)p;     p += (size_t)MTOT * DM * 4;
    _Float16* hn16   = (_Float16*)p;  p += (size_t)MTOT * DM * 2;
    float*    xz     = (float*)p;     p += (size_t)MTOT * 2 * DI * 4;
    float*    xin    = (float*)p;     p += (size_t)MTOT * DI * 4;
    _Float16* xin16  = (_Float16*)p;  p += (size_t)MTOT * DI * 2;
    float*    dbc    = (float*)p;     p += (size_t)MTOT * 64 * 4;
    float*    delta  = (float*)p;     p += (size_t)MTOT * DI * 4;
    float*    yb     = (float*)p;     p += (size_t)MTOT * DI * 4;
    _Float16* y16    = (_Float16*)p;  p += (size_t)MTOT * DI * 2;
    _Float16* w_in16 = (_Float16*)p;  p += (size_t)NLAYER * 2 * DI * DM * 2;
    _Float16* w_xp16 = (_Float16*)p;  p += (size_t)NLAYER * 64 * DI * 2;
    _Float16* w_out16= (_Float16*)p;  p += (size_t)NLAYER * DM * DI * 2;

    // One-shot (per call) weight conversion to f16 for the WMMA GEMMs.
    {
        int n = NLAYER * 2 * DI * DM;
        k_f2h<<<(n + 255) / 256, 256, 0, stream>>>(in_proj_w, w_in16, n);
        n = NLAYER * 64 * DI;
        k_f2h<<<(n + 255) / 256, 256, 0, stream>>>(x_proj_w, w_xp16, n);
        n = NLAYER * DM * DI;
        k_f2h<<<(n + 255) / 256, 256, 0, stream>>>(out_proj_w, w_out16, n);
    }

    k_embed<<<(MTOT * DM) / 256, 256, 0, stream>>>(x, embed_w, embed_b, h);

    for (int l = 0; l < NLAYER; ++l) {
        k_rmsnorm<<<MTOT, 256, 0, stream>>>(h, norm_w + l * DM, hn16);

        dim3 g1(MTOT / 32, (2 * DI) / 64);                 // 4096x2048, K=512
        k_gemm_nt<<<g1, 32, 0, stream>>>(hn16, w_in16 + (size_t)l * 2 * DI * DM,
                                         nullptr, xz, 2 * DI, DM);

        k_conv_silu<<<(MTOT * DI) / 256, 256, 0, stream>>>(xz, conv_w + l * DI * 4,
                                                           conv_b + l * DI, xin, xin16);

        dim3 g2(MTOT / 32, 64 / 64);                       // 4096x64, K=1024
        k_gemm_nt<<<g2, 32, 0, stream>>>(xin16, w_xp16 + (size_t)l * 64 * DI,
                                         nullptr, dbc, 64, DI);

        k_dt<<<(MTOT * DI) / 256, 256, 0, stream>>>(dbc, dt_w + (size_t)l * DI * DTR,
                                                    dt_b + l * DI, delta);

        dim3 g3(B_, DI / 256);
        k_scan<<<g3, 256, 0, stream>>>(dbc, delta, xin, A_log + (size_t)l * DI * DS,
                                       Dvec + l * DI, yb);

        k_gate<<<(MTOT * DI) / 256, 256, 0, stream>>>(xz, yb, y16);

        dim3 g4(MTOT / 32, DM / 64);                       // 4096x512, K=1024 (+residual)
        k_gemm_nt<<<g4, 32, 0, stream>>>(y16, w_out16 + (size_t)l * DM * DI,
                                         h, h, DM, DI);
    }

    k_head<<<(B_ * HZN + 255) / 256, 256, 0, stream>>>(h, head_w, head_b, out);
}
